// TensorGNAN_21964462752475
// MI455X (gfx1250) — compile-verified, matrix operand order checked by reading, not run on hardware
//
#include <hip/hip_runtime.h>

// Problem constants (match reference)
#define N_NODES 2048
#define F_IN    64
#define C_OUT   8
#define H_HID   32
#define G_GR    16
#define J_SPLIT 2      // waves cooperating on one row's j-range

typedef __attribute__((ext_vector_type(16))) _Float16 v16h;
typedef __attribute__((ext_vector_type(8)))  float    v8f;
typedef __attribute__((ext_vector_type(4)))  float    v4f;

// ---------------------------------------------------------------------------
// Kernel 0: zero the [G,C] output (harness poisons it with 0xAA).
// ---------------------------------------------------------------------------
__global__ void zero_out_kernel(float* __restrict__ out, int n) {
    int i = blockIdx.x * blockDim.x + threadIdx.x;
    if (i < n) out[i] = 0.0f;
}

// ---------------------------------------------------------------------------
// Kernel 1: per-row segment bounds (batch sorted -> block-diagonal mask).
// ---------------------------------------------------------------------------
__global__ void row_bounds_kernel(const int* __restrict__ batch,
                                  int* __restrict__ row_lo,
                                  int* __restrict__ row_hi) {
    int i = blockIdx.x * blockDim.x + threadIdx.x;
    if (i >= N_NODES) return;
    int b = batch[i];
    int lo = 0, hi = N_NODES;
    while (lo < hi) { int mid = (lo + hi) >> 1; if (batch[mid] < b) lo = mid + 1; else hi = mid; }
    int lb = lo;
    lo = 0; hi = N_NODES;
    while (lo < hi) { int mid = (lo + hi) >> 1; if (batch[mid] <= b) lo = mid + 1; else hi = mid; }
    row_lo[i] = lb;
    row_hi[i] = lo;
}

// ---------------------------------------------------------------------------
// Kernel 2: f_sum, stored TRANSPOSED and channel-padded: f_sum_T[c][n],
// c = 0..15 (c >= 8 zero) -> WMMA consumer fetches 8 row values per lane
// with two global_load_b128.
// ---------------------------------------------------------------------------
__global__ __launch_bounds__(F_IN)
void fsum_kernel(const float* __restrict__ x,
                 const float* __restrict__ W1,  // [F,H]
                 const float* __restrict__ b1,  // [F,H]
                 const float* __restrict__ W2,  // [F,C,H]
                 const float* __restrict__ b2,  // [F,C]
                 float* __restrict__ f_sum_T)   // [16][N]
{
    __shared__ float s[16];
    const int n = blockIdx.x;
    const int f = threadIdx.x;
    if (f < 16) s[f] = 0.0f;
    __syncthreads();

    const float xv = x[n * F_IN + f];
    float acc[C_OUT];
#pragma unroll
    for (int c = 0; c < C_OUT; ++c) acc[c] = b2[f * C_OUT + c];

#pragma unroll
    for (int h = 0; h < H_HID; ++h) {
        float a = fmaf(xv, W1[f * H_HID + h], b1[f * H_HID + h]);
        a = fmaxf(a, 0.0f);
#pragma unroll
        for (int c = 0; c < C_OUT; ++c)
            acc[c] = fmaf(a, W2[(f * C_OUT + c) * H_HID + h], acc[c]);
    }
#pragma unroll
    for (int c = 0; c < C_OUT; ++c) atomicAdd(&s[c], acc[c]);  // ds_add_f32
    __syncthreads();
    if (f < 16) f_sum_T[f * N_NODES + n] = s[f];
}

// ---------------------------------------------------------------------------
// Per-tile operand bundle for software pipelining.
// ---------------------------------------------------------------------------
struct Tile {
    float d;            // dist[i, j0+m]   (A operand input)
    v4f   nv0, nv1;     // norm[i, jbase..jbase+7]
    v4f   fs0, fs1;     // f_sum_T[c][jbase..jbase+7]
};

__device__ __forceinline__ Tile load_tile(const float* __restrict__ drow,
                                          const float* __restrict__ nrow,
                                          const float* __restrict__ fsc,
                                          int j0, int m, int half) {
    const int jbase = j0 + (half << 3);
    Tile t;
    t.d   = drow[j0 + m];
    t.nv0 = *(const v4f*)(nrow + jbase);
    t.nv1 = *(const v4f*)(nrow + jbase + 4);
    t.fs0 = *(const v4f*)(fsc + jbase);
    t.fs1 = *(const v4f*)(fsc + jbase + 4);
    return t;
}

// ---------------------------------------------------------------------------
// Kernel 3: fused rho-MLP / normalize / mask / contraction / readout.
//
// One wave32 per (row i, j-slice s).  Per 16-j tile:
//   A (16x32 f16): lane(half,m) builds its 16 activations
//       relu(inv_dist*W1[h]+b1[h]) in the ISA A-matrix K order.
//   B (32x16 f16): rho_W2^T zero-padded, VGPR-resident.
//   C (16x16 f32): rho_b2 broadcast  ->  D = A*B + bias in ONE v_wmma.
// Software-pipelined: next tile's 5 loads (1 b32 + 4 b128) issue before the
// current tile's compute, hiding memory latency behind WMMA+VALU.
// Interior tiles take a wave-uniform fast path with no mask compares.
// EXEC stays all-ones throughout (WMMA requirement); loop control is scalar.
// ---------------------------------------------------------------------------
__global__ __launch_bounds__(128)
void rho_wmma_kernel(const float* __restrict__ dist,      // [N,N]
                     const float* __restrict__ norm,      // [N,N]
                     const int*   __restrict__ batch,     // [N]
                     const float* __restrict__ rho_W1,    // [H]
                     const float* __restrict__ rho_b1,    // [H]
                     const float* __restrict__ rho_W2,    // [C,H]
                     const float* __restrict__ rho_b2,    // [C]
                     const float* __restrict__ f_sum_T,   // [16][N]
                     const int*   __restrict__ row_lo,
                     const int*   __restrict__ row_hi,
                     float*       __restrict__ gout)      // [G,C]
{
    const int lane = threadIdx.x;                          // 0..31
    const int wv   = __builtin_amdgcn_readfirstlane(
                         blockIdx.x * blockDim.y + threadIdx.y);
    const int i    = wv / J_SPLIT;                         // row (scalar)
    const int s    = wv % J_SPLIT;                         // j-slice (scalar)
    const int half = lane >> 4;
    const int m    = lane & 15;                            // A row / channel c

    // Per-lane W1/b1 in A-layout K order; resident B operand (rho_W2^T, padded).
    float w1v[16], b1v[16];
    v16h  bmat;
#pragma unroll
    for (int k = 0; k < 16; ++k) {
        const int h = ((k >> 3) << 4) + (half << 3) + (k & 7);
        w1v[k] = rho_W1[h];
        b1v[k] = rho_b1[h];
        bmat[k] = (m < C_OUT) ? (_Float16)rho_W2[m * H_HID + h] : (_Float16)0.0f;
    }
    const float b2c = (m < C_OUT) ? rho_b2[m] : 0.0f;
    v8f cbias;
#pragma unroll
    for (int r = 0; r < 8; ++r) cbias[r] = b2c;            // bias via C operand

    // Wave-uniform segment bounds.
    const int lo = __builtin_amdgcn_readfirstlane(row_lo[i]);
    const int hi = __builtin_amdgcn_readfirstlane(row_hi[i]);
    const int bi = __builtin_amdgcn_readfirstlane(batch[i]);

    const float* __restrict__ drow = dist + (size_t)i * N_NODES;
    const float* __restrict__ nrow = norm + (size_t)i * N_NODES;
    const float* __restrict__ fsc  = f_sum_T + (size_t)m * N_NODES;

    float acc = 0.0f;
    const int start = (lo & ~15) + s * 16;
    const int step  = 16 * J_SPLIT;

    if (start < hi) {
        Tile cur = load_tile(drow, nrow, fsc, start, m, half);   // prologue

        for (int j0 = start; j0 < hi; j0 += step) {
            // ---- issue next tile's loads early (rotating registers) ----
            const int jn = (j0 + step < hi) ? (j0 + step) : j0;  // scalar clamp
            Tile nxt = load_tile(drow, nrow, fsc, jn, m, half);

            const int jbase = j0 + (half << 3);

            // ---- per-row weight w[r] = (mask/norm_safe) * f_sum[j,c] ----
            float w[8];
            if (j0 >= lo && j0 + 16 <= hi) {
                // interior tile: mask is all-ones, skip range compares
#pragma unroll
                for (int r = 0; r < 8; ++r) {
                    const float nv = (r < 4) ? cur.nv0[r] : cur.nv1[r - 4];
                    const float fs = (r < 4) ? cur.fs0[r] : cur.fs1[r - 4];
                    const float rn   = __builtin_amdgcn_rcpf(nv);
                    const float invn = (nv == 0.0f) ? 1.0f : rn;
                    w[r] = invn * fs;
                }
            } else {
                // head/tail tile: apply block-diagonal mask per row
#pragma unroll
                for (int r = 0; r < 8; ++r) {
                    const float nv = (r < 4) ? cur.nv0[r] : cur.nv1[r - 4];
                    const float fs = (r < 4) ? cur.fs0[r] : cur.fs1[r - 4];
                    const float rn   = __builtin_amdgcn_rcpf(nv);
                    const float invn = (nv == 0.0f) ? 1.0f : rn;
                    const int   jr   = jbase + r;
                    const float mv   = (jr >= lo && jr < hi) ? invn : 0.0f;
                    w[r] = mv * fs;
                }
            }

            // ---- build A: 16 f16 activations for this lane's pair ----
            const float inv = __builtin_amdgcn_rcpf(1.0f + cur.d);
            v16h a;
#pragma unroll
            for (int k = 0; k < 16; ++k) {
                float t = fmaf(inv, w1v[k], b1v[k]);
                t = fmaxf(t, 0.0f);
                a[k] = (_Float16)t;
            }

            // ---- D = A*B + bias : 16 pairs x H=32 x 16 ch, one matrix op ----
            v8f d8 = __builtin_amdgcn_wmma_f32_16x16x32_f16(
                false, a, false, bmat, (short)0, cbias, false, false);

            // ---- accumulate: acc += D[r] * w[r]  (8 fmacs) ----
#pragma unroll
            for (int r = 0; r < 8; ++r)
                acc = fmaf(d8[r], w[r], acc);

            cur = nxt;                                     // rotate pipeline
        }
    }

    // combine the two half-wave partials for the same channel
    acc += __shfl_xor(acc, 16, 32);

    // fused segment_sum: graph_out[batch[i], c] += partial
    if (lane < C_OUT)
        atomicAdd(&gout[bi * C_OUT + lane], acc);
}

// ---------------------------------------------------------------------------
// Host side
// ---------------------------------------------------------------------------
extern "C" void kernel_launch(void* const* d_in, const int* in_sizes, int n_in,
                              void* d_out, int out_size, void* d_ws, size_t ws_size,
                              hipStream_t stream) {
    const float* x      = (const float*)d_in[0];
    const float* dist   = (const float*)d_in[1];
    const float* norm   = (const float*)d_in[2];
    const int*   batch  = (const int*)  d_in[3];
    const float* fs_W1  = (const float*)d_in[4];
    const float* fs_b1  = (const float*)d_in[5];
    const float* fs_W2  = (const float*)d_in[6];
    const float* fs_b2  = (const float*)d_in[7];
    const float* rho_W1 = (const float*)d_in[8];
    const float* rho_b1 = (const float*)d_in[9];
    const float* rho_W2 = (const float*)d_in[10];
    const float* rho_b2 = (const float*)d_in[11];
    float* gout = (float*)d_out;

    // workspace layout
    float* f_sum_T = (float*)d_ws;                                 // 16*N floats
    int*   row_lo  = (int*)((char*)d_ws + (size_t)16 * N_NODES * sizeof(float));
    int*   row_hi  = row_lo + N_NODES;

    zero_out_kernel<<<1, 128, 0, stream>>>(gout, out_size);
    row_bounds_kernel<<<(N_NODES + 255) / 256, 256, 0, stream>>>(batch, row_lo, row_hi);
    fsum_kernel<<<N_NODES, F_IN, 0, stream>>>(x, fs_W1, fs_b1, fs_W2, fs_b2, f_sum_T);

    dim3 blk(32, 4);                       // 4 wave32 per block
    const int n_waves = N_NODES * J_SPLIT; // (row, j-slice) per wave
    rho_wmma_kernel<<<n_waves / 4, blk, 0, stream>>>(
        dist, norm, batch, rho_W1, rho_b1, rho_W2, rho_b2,
        f_sum_T, row_lo, row_hi, gout);
}
